// DualStreamEncoderAttention_4234837754550
// MI455X (gfx1250) — matrix operand order repaired and checked by reading, not compiled, not run on hardware
//
#include <hip/hip_runtime.h>
#include <hip/hip_bf16.h>

// ---------------------------------------------------------------------------
// DualStreamEncoderAttention for MI455X (gfx1250, wave32, WMMA)
// B=4, S=1024, H=1024, heads=16, head_dim=64, concat-KV length 2048
// ---------------------------------------------------------------------------

typedef __attribute__((ext_vector_type(16))) _Float16 v16h;
typedef __attribute__((ext_vector_type(8)))  _Float16 v8h;
typedef __attribute__((ext_vector_type(8)))  float    v8f;

union Frag16 {
    v16h v;
    v8h  h[2];
    _Float16 e[16];
};

#define BATCH  4
#define SEQ    1024
#define HID    1024
#define NHEAD  16
#define HDIM   64
#define ROWS   (BATCH * SEQ)          // 4096
#define QKVN   (3 * HID)              // 3072
#define KVLEN  (2 * SEQ)              // 2048
#define LOG2E  1.44269504088896f

// ---------------------------------------------------------------------------
// 1) RoPE tables: ctab/stab [1024][64] fp32 (pos = y*32 + x)
// ---------------------------------------------------------------------------
__global__ __launch_bounds__(256) void rope_tab_kernel(float* __restrict__ ctab,
                                                       float* __restrict__ stab) {
    int idx = blockIdx.x * 256 + threadIdx.x;     // 65536 total
    int pos = idx >> 6;
    int d   = idx & 63;
    float fpos = (d < 32) ? (float)(pos & 31) : (float)(pos >> 5);
    int   i    = (d & 31) & 15;                   // (d % 32) % 16
    float inv  = powf(10000.0f, -((float)(2 * i)) / 32.0f);
    float ang  = fpos * inv;
    ctab[idx] = cosf(ang);
    stab[idx] = sinf(ang);
}

// ---------------------------------------------------------------------------
// 2) LayerNorm (fp32) -> f16 cast; both streams via blockIdx.y
// ---------------------------------------------------------------------------
__global__ __launch_bounds__(256) void ln_cast_kernel(const float* __restrict__ xa,
                                                      const float* __restrict__ xb,
                                                      const float* __restrict__ ga,
                                                      const float* __restrict__ ba,
                                                      const float* __restrict__ gb,
                                                      const float* __restrict__ bb,
                                                      _Float16* __restrict__ xln) {
    const int row  = blockIdx.x;                  // 0..4095
    const int strm = blockIdx.y;                  // 0 or 1
    const float* x  = (strm ? xb : xa) + (size_t)row * HID;
    const float* g  = strm ? gb : ga;
    const float* be = strm ? bb : ba;

    __shared__ float red[256];
    float v[4];
    float sum = 0.0f;
#pragma unroll
    for (int i = 0; i < 4; ++i) { v[i] = x[threadIdx.x + i * 256]; sum += v[i]; }
    red[threadIdx.x] = sum;
    __syncthreads();
    for (int st = 128; st > 0; st >>= 1) {
        if ((int)threadIdx.x < st) red[threadIdx.x] += red[threadIdx.x + st];
        __syncthreads();
    }
    float mu = red[0] * (1.0f / (float)HID);
    __syncthreads();
    float vs = 0.0f;
#pragma unroll
    for (int i = 0; i < 4; ++i) { float d = v[i] - mu; vs += d * d; }
    red[threadIdx.x] = vs;
    __syncthreads();
    for (int st = 128; st > 0; st >>= 1) {
        if ((int)threadIdx.x < st) red[threadIdx.x] += red[threadIdx.x + st];
        __syncthreads();
    }
    float rstd = rsqrtf(red[0] * (1.0f / (float)HID) + 1e-5f);

    _Float16* o = xln + (size_t)strm * ROWS * HID + (size_t)row * HID;
#pragma unroll
    for (int i = 0; i < 4; ++i) {
        int col = threadIdx.x + i * 256;
        o[col] = (_Float16)((v[i] - mu) * rstd * g[col] + be[col]);
    }
}

// ---------------------------------------------------------------------------
// 3) Weight cast + transpose: w [K][N] fp32 -> wT [N][K] f16 (K contiguous)
// ---------------------------------------------------------------------------
__global__ __launch_bounds__(256) void wcastT_kernel(const float* __restrict__ w,
                                                     _Float16* __restrict__ wT,
                                                     int K, int N) {
    size_t idx = (size_t)blockIdx.x * 256 + threadIdx.x;
    size_t total = (size_t)K * N;
    if (idx >= total) return;
    int n = (int)(idx / K);
    int k = (int)(idx % K);
    wT[idx] = (_Float16)w[(size_t)k * N + n];
}

// ---------------------------------------------------------------------------
// 4) QKV GEMM (WMMA f16) with fused RoPE + scatter epilogue.
//    A = xln[strm] [4096][1024], Bt = wqkvT[strm] [3072][1024]
//    Wave computes a 64x64 tile (head-aligned), 4 waves per WG -> 128x128.
// ---------------------------------------------------------------------------
__global__ __launch_bounds__(128) void qkv_gemm_rope_kernel(
        const _Float16* __restrict__ xln,
        const _Float16* __restrict__ wqkvT,
        const float* __restrict__ ctab,
        const float* __restrict__ stab,
        _Float16* __restrict__ qT,   // [2][B][16][S][64]
        _Float16* __restrict__ kc,   // [B][16][2048][64]
        _Float16* __restrict__ vt)   // [B][16][64][2048]
{
    const int lane = threadIdx.x & 31;
    const int lm   = lane & 15;
    const int half = lane >> 4;
    const int w    = threadIdx.x >> 5;
    const int strm = blockIdx.z;
    const int m0   = blockIdx.y * 128 + (w >> 1) * 64;
    const int n0   = blockIdx.x * 128 + (w & 1) * 64;

    const _Float16* A  = xln   + (size_t)strm * ROWS * HID;
    const _Float16* Bt = wqkvT + (size_t)strm * QKVN * HID;

    v8f acc[4][4] = {};

    for (int kk = 0; kk < HID; kk += 32) {
        Frag16 a[4], b[4];
#pragma unroll
        for (int i = 0; i < 4; ++i) {
            const _Float16* ap = A + (size_t)(m0 + i * 16 + lm) * HID + kk;
            const v8h* ar = (const v8h*)ap;
            a[i].h[0] = ar[half];        // K = kk + half*8 .. +7
            a[i].h[1] = ar[2 + half];    // K = kk + 16 + half*8 .. +7
            if (kk + 32 < HID) __builtin_prefetch(ap + 32, 0, 3);
        }
#pragma unroll
        for (int j = 0; j < 4; ++j) {
            const _Float16* bp = Bt + (size_t)(n0 + j * 16 + lm) * HID + kk;
            const v8h* br = (const v8h*)bp;
            b[j].h[0] = br[half];          // K = kk + half*8 .. +7
            b[j].h[1] = br[2 + half];      // K = kk + 16 + half*8 .. +7
        }
#pragma unroll
        for (int i = 0; i < 4; ++i)
#pragma unroll
            for (int j = 0; j < 4; ++j)
                acc[i][j] = __builtin_amdgcn_wmma_f32_16x16x32_f16(
                    false, a[i].v, false, b[j].v, (short)0, acc[i][j], false, false);
    }

    // Epilogue: wave's 64 columns == exactly one (which, head), d = j*16 + lm.
    const int which = n0 >> 10;            // 0=q 1=k 2=v
    const int head  = (n0 & 1023) >> 6;
#pragma unroll
    for (int i = 0; i < 4; ++i) {
#pragma unroll
        for (int r = 0; r < 8; ++r) {
            int mrow = m0 + i * 16 + r + half * 8;
            int bidx = mrow >> 10;
            int s    = mrow & 1023;
#pragma unroll
            for (int j = 0; j < 4; ++j) {
                int   d = j * 16 + lm;
                float x = acc[i][j][r];
                if (which == 2) {
                    vt[(((size_t)(bidx * NHEAD + head) * HDIM + d) * KVLEN)
                       + strm * SEQ + s] = (_Float16)x;
                } else {
                    float partner = acc[i][j ^ 2][r];           // d +/- 32
                    float rot = (j < 2) ? -partner : partner;
                    float c  = ctab[s * 64 + d];
                    float sn = stab[s * 64 + d];
                    float val = x * c + rot * sn;
                    if (which == 0)
                        qT[((((size_t)(strm * BATCH + bidx) * NHEAD + head) * SEQ) + s) * HDIM + d]
                            = (_Float16)val;
                    else
                        kc[(((size_t)(bidx * NHEAD + head) * KVLEN) + strm * SEQ + s) * HDIM + d]
                            = (_Float16)val;
                }
            }
        }
    }
}

// ---------------------------------------------------------------------------
// 5) Flash attention. One wave = 16 queries; WG = 4 waves = 64 queries.
//    Scores computed transposed (S^T = K*Q^T) so query lives in the lane,
//    making row max/sum a lane-local reduction + one shfl_xor(16).
//    P fragment re-packs into A-layout with NO cross-lane movement.
// ---------------------------------------------------------------------------
__global__ __launch_bounds__(128) void attention_kernel(
        const _Float16* __restrict__ qT,
        const _Float16* __restrict__ kc,
        const _Float16* __restrict__ vt,
        _Float16* __restrict__ attnb)   // [2][4096][1024]
{
    const int lane = threadIdx.x & 31;
    const int lm   = lane & 15;
    const int half = lane >> 4;
    const int w    = threadIdx.x >> 5;
    const int strm = blockIdx.z >> 2;
    const int b    = blockIdx.z & 3;
    const int h    = blockIdx.y;
    const int qbase = blockIdx.x * 64 + w * 16;
    const float scale = 0.125f;          // 1/sqrt(64)

    // Q^T B-fragments, held for the whole loop (lane = query).
    const _Float16* qp = qT + ((((size_t)(strm * BATCH + b) * NHEAD + h) * SEQ) + qbase + lm) * HDIM;
    Frag16 bq[2];
#pragma unroll
    for (int f = 0; f < 2; ++f) {
        const v8h* qr = (const v8h*)(qp + f * 32 + half * 8);
        bq[f].h[0] = qr[0];
        bq[f].h[1] = qr[2];
    }

    const _Float16* kbase = kc + (size_t)(b * NHEAD + h) * KVLEN * HDIM;
    const _Float16* vbase = vt + (size_t)(b * NHEAD + h) * HDIM * KVLEN;

    float mi = -3.0e38f, li = 0.0f;
    v8f o[4] = {};
    const v8f vzero = {};

    for (int kb = 0; kb < KVLEN; kb += 32) {
        // ---- scores^T for two 16-key chunks ----
        v8f sc[2];
#pragma unroll
        for (int c = 0; c < 2; ++c) {
            Frag16 ak[2];
            const _Float16* kp = kbase + (size_t)(kb + c * 16 + lm) * HDIM;
#pragma unroll
            for (int f = 0; f < 2; ++f) {
                const v8h* kr = (const v8h*)(kp + f * 32);
                ak[f].h[0] = kr[half];       // d = f*32 + half*8
                ak[f].h[1] = kr[2 + half];   // d = f*32 + 16 + half*8
            }
            v8f s8 = vzero;
            s8 = __builtin_amdgcn_wmma_f32_16x16x32_f16(false, ak[0].v, false, bq[0].v,
                                                        (short)0, s8, false, false);
            s8 = __builtin_amdgcn_wmma_f32_16x16x32_f16(false, ak[1].v, false, bq[1].v,
                                                        (short)0, s8, false, false);
            sc[c] = s8;
        }
        // ---- online softmax (lane = query) ----
        float rmax = -3.0e38f;
#pragma unroll
        for (int c = 0; c < 2; ++c)
#pragma unroll
            for (int r = 0; r < 8; ++r) {
                float vv = sc[c][r] * scale;
                sc[c][r] = vv;
                rmax = fmaxf(rmax, vv);
            }
        rmax = fmaxf(rmax, __shfl_xor(rmax, 16, 32));
        float mnew = fmaxf(mi, rmax);
        float rsum = 0.0f;
        Frag16 ap;
#pragma unroll
        for (int c = 0; c < 2; ++c)
#pragma unroll
            for (int r = 0; r < 8; ++r) {
                float p = __builtin_amdgcn_exp2f((sc[c][r] - mnew) * LOG2E);
                rsum += p;
                ap.e[c * 8 + r] = (_Float16)p;   // lands exactly in A-layout
            }
        rsum += __shfl_xor(rsum, 16, 32);
        float corr = __builtin_amdgcn_exp2f((mi - mnew) * LOG2E);
        li = li * corr + rsum;
        mi = mnew;
        // ---- rescale O (query = vgpr r + 8*half in C-layout) ----
#pragma unroll
        for (int r = 0; r < 8; ++r) {
            float cb = __shfl(corr, r + half * 8, 32);
#pragma unroll
            for (int t = 0; t < 4; ++t) o[t][r] *= cb;
        }
        // ---- O += P * V (V^T rows contiguous in keys) ----
#pragma unroll
        for (int t = 0; t < 4; ++t) {
            Frag16 bv;
            const v8h* vr = (const v8h*)(vbase + (size_t)(t * 16 + lm) * KVLEN + kb + half * 8);
            bv.h[0] = vr[0];
            bv.h[1] = vr[2];
            o[t] = __builtin_amdgcn_wmma_f32_16x16x32_f16(false, ap.v, false, bv.v,
                                                          (short)0, o[t], false, false);
        }
    }
    // ---- normalize + store f16 [strm][b*1024+s][h*64+d] ----
    float inv = 1.0f / li;
#pragma unroll
    for (int r = 0; r < 8; ++r) {
        float ib = __shfl(inv, r + half * 8, 32);
        int mrow = qbase + r + half * 8;
#pragma unroll
        for (int t = 0; t < 4; ++t) {
            attnb[(size_t)strm * ROWS * HID + (size_t)(b * SEQ + mrow) * HID + h * HDIM + t * 16 + lm]
                = (_Float16)(o[t][r] * ib);
        }
    }
}

// ---------------------------------------------------------------------------
// 6) Output projection GEMM (WMMA f16) -> fp32 d_out, tuple-concat order.
// ---------------------------------------------------------------------------
__global__ __launch_bounds__(128) void out_gemm_kernel(
        const _Float16* __restrict__ attnb,
        const _Float16* __restrict__ woutT,
        float* __restrict__ out)
{
    const int lane = threadIdx.x & 31;
    const int lm   = lane & 15;
    const int half = lane >> 4;
    const int w    = threadIdx.x >> 5;
    const int strm = blockIdx.z;
    const int m0   = blockIdx.y * 128 + (w >> 1) * 64;
    const int n0   = blockIdx.x * 128 + (w & 1) * 64;

    const _Float16* A  = attnb + (size_t)strm * ROWS * HID;
    const _Float16* Bt = woutT + (size_t)strm * HID * HID;

    v8f acc[4][4] = {};

    for (int kk = 0; kk < HID; kk += 32) {
        Frag16 a[4], b[4];
#pragma unroll
        for (int i = 0; i < 4; ++i) {
            const _Float16* ap = A + (size_t)(m0 + i * 16 + lm) * HID + kk;
            const v8h* ar = (const v8h*)ap;
            a[i].h[0] = ar[half];
            a[i].h[1] = ar[2 + half];
            if (kk + 32 < HID) __builtin_prefetch(ap + 32, 0, 3);
        }
#pragma unroll
        for (int j = 0; j < 4; ++j) {
            const _Float16* bp = Bt + (size_t)(n0 + j * 16 + lm) * HID + kk;
            const v8h* br = (const v8h*)bp;
            b[j].h[0] = br[half];
            b[j].h[1] = br[2 + half];
        }
#pragma unroll
        for (int i = 0; i < 4; ++i)
#pragma unroll
            for (int j = 0; j < 4; ++j)
                acc[i][j] = __builtin_amdgcn_wmma_f32_16x16x32_f16(
                    false, a[i].v, false, b[j].v, (short)0, acc[i][j], false, false);
    }

    float* op = out + (size_t)strm * ROWS * HID;
#pragma unroll
    for (int i = 0; i < 4; ++i)
#pragma unroll
        for (int r = 0; r < 8; ++r) {
            int mrow = m0 + i * 16 + r + half * 8;
#pragma unroll
            for (int j = 0; j < 4; ++j)
                op[(size_t)mrow * HID + n0 + j * 16 + lm] = acc[i][j][r];
        }
}

// ---------------------------------------------------------------------------
// Host-side launch
// ---------------------------------------------------------------------------
extern "C" void kernel_launch(void* const* d_in, const int* in_sizes, int n_in,
                              void* d_out, int out_size, void* d_ws, size_t ws_size,
                              hipStream_t stream) {
    const float* x_a    = (const float*)d_in[0];
    const float* x_b    = (const float*)d_in[1];
    const float* Wqkv_a = (const float*)d_in[2];
    const float* Wqkv_b = (const float*)d_in[3];
    const float* Wout_a = (const float*)d_in[4];
    const float* Wout_b = (const float*)d_in[5];
    const float* g_a    = (const float*)d_in[6];
    const float* b_a    = (const float*)d_in[7];
    const float* g_b    = (const float*)d_in[8];
    const float* b_b    = (const float*)d_in[9];
    (void)in_sizes; (void)n_in; (void)out_size; (void)ws_size;

    // Workspace carve-up (f16 halves unless noted), total ~96.5 MB.
    _Float16* xln   = (_Float16*)d_ws;                          // 2*4096*1024
    _Float16* wqkvT = xln   + (size_t)2 * ROWS * HID;           // 2*3072*1024
    _Float16* woutT = wqkvT + (size_t)2 * QKVN * HID;           // 2*1024*1024
    _Float16* qTp   = woutT + (size_t)2 * HID  * HID;           // 2*4*16*1024*64
    _Float16* kcp   = qTp   + (size_t)2 * BATCH * NHEAD * SEQ * HDIM;   // 4*16*2048*64
    _Float16* vtp   = kcp   + (size_t)BATCH * NHEAD * KVLEN * HDIM;     // 4*16*64*2048
    _Float16* attnb = vtp   + (size_t)BATCH * NHEAD * HDIM * KVLEN;     // 2*4096*1024
    float*    ctab  = (float*)(attnb + (size_t)2 * ROWS * HID);         // 1024*64 f32
    float*    stab  = ctab + SEQ * HDIM;

    rope_tab_kernel<<<dim3((SEQ * HDIM) / 256), 256, 0, stream>>>(ctab, stab);

    ln_cast_kernel<<<dim3(ROWS, 2), 256, 0, stream>>>(x_a, x_b, g_a, b_a, g_b, b_b, xln);

    wcastT_kernel<<<dim3((QKVN * HID) / 256), 256, 0, stream>>>(Wqkv_a, wqkvT, HID, QKVN);
    wcastT_kernel<<<dim3((QKVN * HID) / 256), 256, 0, stream>>>(Wqkv_b, wqkvT + (size_t)QKVN * HID, HID, QKVN);
    wcastT_kernel<<<dim3((HID * HID) / 256), 256, 0, stream>>>(Wout_a, woutT, HID, HID);
    wcastT_kernel<<<dim3((HID * HID) / 256), 256, 0, stream>>>(Wout_b, woutT + (size_t)HID * HID, HID, HID);

    qkv_gemm_rope_kernel<<<dim3(QKVN / 128, ROWS / 128, 2), 128, 0, stream>>>(
        xln, wqkvT, ctab, stab, qTp, kcp, vtp);

    attention_kernel<<<dim3(SEQ / 64, NHEAD, 2 * BATCH), 128, 0, stream>>>(
        qTp, kcp, vtp, attnb);

    out_gemm_kernel<<<dim3(HID / 128, ROWS / 128, 2), 128, 0, stream>>>(
        attnb, woutT, (float*)d_out);
}
